// DenseMotionNetwork_40450001993818
// MI455X (gfx1250) — matrix operand order repaired and checked by reading, not compiled, not run on hardware
//
#include <hip/hip_runtime.h>
#include <cstdint>
#include <cstddef>

typedef __bf16 bf16;
typedef __bf16 v16bf __attribute__((ext_vector_type(16)));
typedef float  v8f   __attribute__((ext_vector_type(8)));

#define DEV_INLINE __device__ __forceinline__

// ---------------- problem constants ----------------
static constexpr int BSZ = 8, FEAT_CH = 32, DDIM = 16, HDIM = 64, WDIM = 64;
static constexpr int DHW = DDIM * HDIM * WDIM;            // 65536 per batch
static constexpr long N_FULL = (long)BSZ * DHW;           // 524288 voxels

// ======================================================================
// Weight repack: f32 [Co][Ci][KS][KS][KS] -> bf16 [CoPad][Kpad],
// k = ((kd*KS+kh)*KS+kw)*Ci + ci  (spatial-major so K is Ci-contiguous,
// matching the channel-last activation layout), zero padded.
// ======================================================================
__global__ void repack_w_kernel(const float* __restrict__ w, bf16* __restrict__ wp,
                                int Co, int Ci, int KS, int CoPad, int Kpad, int Ktot)
{
    long i = (long)blockIdx.x * 256 + threadIdx.x;
    long total = (long)CoPad * Kpad;
    if (i >= total) return;
    int k  = (int)(i % Kpad);
    int co = (int)(i / Kpad);
    float v = 0.f;
    if (co < Co && k < Ktot) {
        int ci = k % Ci;
        int sp = k / Ci;
        int kw = sp % KS; int t = sp / KS;
        int kh = t % KS;  int kd = t / KS;
        v = w[((((size_t)co * Ci + ci) * KS + kd) * KS + kh) * KS + kw];
    }
    wp[i] = (bf16)v;
}

// NCDHW f32 -> channel-last bf16 [b][s][c]
__global__ void to_cl_bf16_kernel(const float* __restrict__ src, bf16* __restrict__ dst,
                                  int Bn, int C, long S)
{
    long i = (long)blockIdx.x * 256 + threadIdx.x;
    long total = (long)Bn * C * S;
    if (i >= total) return;
    int  c = (int)(i % C);
    long n = i / C;
    long s = n % S;
    int  b = (int)(n / S);
    dst[i] = (bf16)src[((size_t)b * C + c) * S + s];
}

__global__ void zero_f32_kernel(float* __restrict__ p, int n)
{
    int i = blockIdx.x * 256 + threadIdx.x;
    if (i < n) p[i] = 0.f;
}

// ======================================================================
// Implicit-GEMM conv3d with v_wmma_f32_16x16x32_bf16 (wave32).
//   block tile : 32 channels (M) x 128 voxels (N), K-step 64
//   8 waves    : wave = (msub in 0..1) x (npair in 0..3); each wave owns
//                two 16x16 D tiles sharing one A fragment -> 4 WMMAs/body
//   A slab     : staged with GLOBAL_LOAD_ASYNC_TO_LDS_B128 (ASYNCcnt)
//   B slab     : im2col gather, incremental (sp,ci) walk, 32-bit address
//                math, register-packed b128 LDS stores
//   pipelining : ping-pong LDS slabs -> single barrier per K-step
//   epilogue   : bf16 channel-last store + per-channel sum/sumsq atomics
//                for the following train-mode batch-norm
//   ups=1 folds the 1x2x2 nearest upsample into the gather.
// ======================================================================
template <int KS>
__global__ __launch_bounds__(256)
void conv3d_wmma_kernel(const bf16* __restrict__ X, const bf16* __restrict__ Wt,
                        bf16* __restrict__ Y, float* __restrict__ stats,
                        int D, int H, int W,
                        int Ci, int CoStore, int CoPad,
                        int Kpad, int ups, int spStep, int ciStep)
{
    constexpr int PAD = KS / 2;
    constexpr int SPT = KS * KS * KS;

    __shared__ __attribute__((aligned(16))) bf16 As[2][32][72];   // ping-pong 32M x 64K
    __shared__ __attribute__((aligned(16))) bf16 Bs[2][128][72];  // ping-pong 128N x 64K
    constexpr unsigned ASLAB = 32 * 72 * 2;                       // bytes per A slab

    const int tid   = threadIdx.x;
    const int lane  = tid & 31;
    const int wv    = tid >> 5;
    const int lr    = lane & 15;
    const int lh    = lane >> 4;
    const int msub  = wv & 1;      // 2 M sub-tiles
    const int npair = wv >> 1;     // 4 N tile-pairs

    const long nBase = (long)blockIdx.x * 128;
    const int  mBase = blockIdx.y * 32;
    const int  Hin = ups ? (H >> 1) : H;
    const int  Win = ups ? (W >> 1) : W;

    // ---- B-gather assignment: voxel nl, 32 contiguous k at kq ----
    const int nl = tid >> 1;
    const int kq = (tid & 1) * 32;
    long n = nBase + nl;
    int w_ = (int)(n % W); long t = n / W;
    int h_ = (int)(t % H); t /= H;
    int d_ = (int)(t % D);
    int b_ = (int)(t / D);

    int sp = kq / Ci;              // single division per thread
    int ci = kq - sp * Ci;

    // ---- A staging: channel mA, 8 contiguous k at kA (16 bytes) ----
    const int mA = tid >> 3;
    const int kA = (tid & 7) * 8;
    const bf16* wrow = Wt + (size_t)(mBase + mA) * Kpad + kA;
    const unsigned ldsA0 = (unsigned)(uintptr_t)&As[0][mA][kA];

    v8f zero8 = {0.f, 0.f, 0.f, 0.f, 0.f, 0.f, 0.f, 0.f};
    v8f accs[2] = {zero8, zero8};

    for (int k0 = 0; k0 < Kpad; k0 += 64) {
        const int pb = (k0 >> 6) & 1;   // ping-pong slab

        // --- async weight stage (ASYNCcnt) ---
        uint64_t ga = (uint64_t)(uintptr_t)(wrow + k0);
        unsigned ldsA = ldsA0 + (unsigned)pb * ASLAB;
        asm volatile("global_load_async_to_lds_b128 %0, %1, off"
                     :: "v"(ldsA), "v"(ga) : "memory");

        // --- activation gather: incremental (sp,ci) walk, 32-bit math ---
        int spE = sp, ciE = ci;
        const bf16* xp = X;
        bool ok = false;
        auto decodeSp = [&](int s) {
            int kw = s % KS; int t2 = s / KS;
            int kh = t2 % KS; int kd = t2 / KS;
            int id = d_ + kd - PAD;
            int ih = h_ + kh - PAD;
            int iw = w_ + kw - PAD;
            ok = (s < SPT) && ((unsigned)id < (unsigned)D) &&
                 ((unsigned)ih < (unsigned)H) && ((unsigned)iw < (unsigned)W);
            if (ok) {
                int sh = ups ? (ih >> 1) : ih;
                int sw = ups ? (iw >> 1) : iw;
                int base = (((b_ * D + id) * Hin + sh) * Win + sw) * Ci;  // < 2^31
                xp = X + base;
            }
        };
        decodeSp(spE);
        #pragma unroll
        for (int j = 0; j < 4; ++j) {
            union { bf16 h[8]; uint4 u; } pk;
            #pragma unroll
            for (int e = 0; e < 8; ++e) {
                pk.h[e] = ok ? xp[ciE] : (bf16)0.f;
                if (++ciE == Ci) { ciE = 0; decodeSp(++spE); }
            }
            *(uint4*)&Bs[pb][nl][kq + 8 * j] = pk.u;   // vector ds_store
        }
        // advance (sp,ci) by 64 for next K-step (host precomputed 64/Ci, 64%Ci)
        sp += spStep; ci += ciStep;
        if (ci >= Ci) { ci -= Ci; ++sp; }

        asm volatile("s_wait_asynccnt 0" ::: "memory");
        __syncthreads();   // single barrier per K-step (ping-pong slabs)

        // --- 2 K-chunks x (1 A frag, 2 B frags) -> 4 WMMAs ---
        const bf16* ar  = &As[pb][16 * msub + lr][0];
        const bf16* br0 = &Bs[pb][16 * (2 * npair + 0) + lr][0];
        const bf16* br1 = &Bs[pb][16 * (2 * npair + 1) + lr][0];
        #pragma unroll
        for (int c = 0; c < 2; ++c) {
            v16bf a, b0, b1;
            #pragma unroll
            for (int e = 0; e < 8; ++e) {
                a[e]     = ar[32 * c + 16 * lh + e];       // K = 16*lh + e
                a[8 + e] = ar[32 * c + 16 * lh + 8 + e];   // K = 16*lh + 8 + e
            }
            #pragma unroll
            for (int e = 0; e < 16; ++e) {
                b0[e] = br0[32 * c + 16 * lh + e];          // K = 16*lh + e
                b1[e] = br1[32 * c + 16 * lh + e];
            }
            accs[0] = __builtin_amdgcn_wmma_f32_16x16x32_bf16(false, a, false, b0,
                                                              (short)0, accs[0], false, false);
            accs[1] = __builtin_amdgcn_wmma_f32_16x16x32_bf16(false, a, false, b1,
                                                              (short)0, accs[1], false, false);
        }
    }

    // ---- epilogue: store + BN statistics ----
    const int coBase = mBase + 16 * msub + 8 * lh;
    #pragma unroll
    for (int tN = 0; tN < 2; ++tN) {
        long nOut = nBase + 16 * (2 * npair + tN) + lr;
        size_t yb = (size_t)nOut * CoStore + coBase;
        if ((CoStore & 7) == 0) {
            union { bf16 h[8]; uint4 u; } pk;
            #pragma unroll
            for (int v = 0; v < 8; ++v) pk.h[v] = (bf16)accs[tN][v];
            *(uint4*)(Y + yb) = pk.u;               // 16-byte store
        } else {
            #pragma unroll
            for (int v = 0; v < 8; ++v)
                if (coBase + v < CoStore) Y[yb + v] = (bf16)accs[tN][v];
        }
    }
    if (stats) {
        #pragma unroll
        for (int v = 0; v < 8; ++v) {
            float s = accs[0][v] + accs[1][v];
            float q = accs[0][v] * accs[0][v] + accs[1][v] * accs[1][v];
            #pragma unroll
            for (int off = 1; off < 16; off <<= 1) {
                s += __shfl_xor(s, off, 32);
                q += __shfl_xor(q, off, 32);
            }
            if (lr == 0 && coBase + v < CoStore) {
                atomicAdd(&stats[coBase + v], s);
                atomicAdd(&stats[CoPad + coBase + v], q);
            }
        }
    }
}

// ======================================================================
// BN(train-mode) + ReLU, fused with 1x2x2 avg-pool (down path) or a
// channel-offset write into a concat buffer (up path). bf16 channel-last.
// ======================================================================
__global__ void bn_apply_kernel(const bf16* __restrict__ Yc, const float* __restrict__ stats,
                                const float* __restrict__ g, const float* __restrict__ be,
                                bf16* __restrict__ dst,
                                int B, int D, int H, int W,
                                int Co, int CoPad, int Cdst, int coff, int pool)
{
    int Ho = pool ? (H >> 1) : H;
    int Wo = pool ? (W >> 1) : W;
    long total = (long)B * D * Ho * Wo * Co;
    long i = (long)blockIdx.x * 256 + threadIdx.x;
    if (i >= total) return;
    int  c = (int)(i % Co);
    long n = i / Co;
    int wo = (int)(n % Wo); long t = n / Wo;
    int ho = (int)(t % Ho); t /= Ho;
    int d  = (int)(t % D);
    int b  = (int)(t / D);

    float invN  = 1.0f / ((float)B * D * H * W);
    float mean  = stats[c] * invN;
    float var   = stats[CoPad + c] * invN - mean * mean;
    float scale = g[c] * rsqrtf(var + 1e-5f);
    float shift = be[c] - mean * scale;

    float val;
    if (pool) {
        float s = 0.f;
        #pragma unroll
        for (int dy = 0; dy < 2; ++dy)
            #pragma unroll
            for (int dx = 0; dx < 2; ++dx) {
                size_t p = ((((size_t)b * D + d) * H + 2 * ho + dy) * W + 2 * wo + dx) * Co + c;
                float x = (float)Yc[p] * scale + shift;
                s += fmaxf(x, 0.f);
            }
        val = s * 0.25f;
    } else {
        size_t p = ((((size_t)b * D + d) * H + ho) * W + wo) * Co + c;
        float x = (float)Yc[p] * scale + shift;
        val = fmaxf(x, 0.f);
    }
    dst[(((size_t)b * D + d) * Ho * Wo + (size_t)ho * Wo + wo) * Cdst + coff + c] = (bf16)val;
}

// skip-connection concat copy
__global__ void copy_ch_kernel(const bf16* __restrict__ src, bf16* __restrict__ dst,
                               long N, int Csrc, int Cdst, int coff)
{
    long i = (long)blockIdx.x * 256 + threadIdx.x;
    long total = N * Csrc;
    if (i >= total) return;
    int  c = (int)(i % Csrc);
    long n = i / Csrc;
    dst[n * Cdst + coff + c] = src[i];
}

// ======================================================================
// Keypoint / jacobian setup (tiny): one lane per batch element.
// ======================================================================
DEV_INLINE void inv3x3(const float* a, float* o)
{
    float a00=a[0],a01=a[1],a02=a[2],a10=a[3],a11=a[4],a12=a[5],a20=a[6],a21=a[7],a22=a[8];
    float c00 =  a11*a22 - a12*a21;
    float c10 = -(a10*a22 - a12*a20);
    float c20 =  a10*a21 - a11*a20;
    float det = a00*c00 + a01*c10 + a02*c20;
    float id  = 1.f / det;
    o[0]=c00*id;                 o[1]=-(a01*a22-a02*a21)*id;  o[2]= (a01*a12-a02*a11)*id;
    o[3]=c10*id;                 o[4]= (a00*a22-a02*a20)*id;  o[5]=-(a00*a12-a02*a10)*id;
    o[6]=c20*id;                 o[7]=-(a00*a21-a01*a20)*id;  o[8]= (a00*a11-a01*a10)*id;
}

__global__ void kp_setup_kernel(const float* __restrict__ mesh,
                                const float* __restrict__ srcR, const float* __restrict__ srcC,
                                const float* __restrict__ srcT,
                                const float* __restrict__ drvR, const float* __restrict__ drvC,
                                const float* __restrict__ drvT,
                                float* __restrict__ kpb)  // [8][3] kp_src, [8][3] kp_drv, [8][9] jac
{
    int b = threadIdx.x;
    if (b >= BSZ) return;
    const float* m = mesh + (size_t)b * 478 * 3;
    float mnx = m[0], mxx = m[0], mny = m[1], mxy = m[1], mxz = m[2];
    for (int i = 1; i < 478; ++i) {
        float x = m[i*3], y = m[i*3+1], z = m[i*3+2];
        mnx = fminf(mnx, x); mxx = fmaxf(mxx, x);
        mny = fminf(mny, y); mxy = fmaxf(mxy, y);
        mxz = fmaxf(mxz, z);
    }
    float ctr[3] = {0.5f*(mxx+mnx), 0.5f*(mxy+mny), mxz};

    float Si[9], Di[9];
    inv3x3(srcR + b*9, Si);
    inv3x3(drvR + b*9, Di);
    float cs = srcC[b], cd = drvC[b];

    for (int j = 0; j < 3; ++j) {
        float ss = 0.f, dd = 0.f;
        for (int k = 0; k < 3; ++k) {
            ss += Si[j*3+k] * (ctr[k] - srcT[b*3+k]);
            dd += Di[j*3+k] * (ctr[k] - drvT[b*3+k]);
        }
        kpb[b*3 + j]      = ss / cs;   // kp_src
        kpb[24 + b*3 + j] = dd / cd;   // kp_drv
    }
    float r = cd / cs;
    for (int i = 0; i < 3; ++i)
        for (int j = 0; j < 3; ++j) {
            float s = 0.f;
            for (int k = 0; k < 3; ++k) s += drvR[b*9 + i*3 + k] * Si[k*3 + j];
            kpb[48 + b*9 + i*3 + j] = r * s;   // jac
        }
}

// ======================================================================
// Sparse motion + heatmap + trilinear grid-sample of compressed features.
// ======================================================================
DEV_INLINE void trisample(const bf16* __restrict__ fb, float x, float y, float z, float* o4)
{
    float ix = ((x + 1.f) * (float)WDIM - 1.f) * 0.5f;
    float iy = ((y + 1.f) * (float)HDIM - 1.f) * 0.5f;
    float iz = ((z + 1.f) * (float)DDIM - 1.f) * 0.5f;
    float x0 = floorf(ix), y0 = floorf(iy), z0 = floorf(iz);
    float wx1 = ix - x0, wy1 = iy - y0, wz1 = iz - z0;
    o4[0] = o4[1] = o4[2] = o4[3] = 0.f;
    #pragma unroll
    for (int cz = 0; cz < 2; ++cz)
        #pragma unroll
        for (int cy = 0; cy < 2; ++cy)
            #pragma unroll
            for (int cx = 0; cx < 2; ++cx) {
                int xi = (int)x0 + cx, yi = (int)y0 + cy, zi = (int)z0 + cz;
                float wgt = (cx ? wx1 : 1.f - wx1) * (cy ? wy1 : 1.f - wy1) * (cz ? wz1 : 1.f - wz1);
                if (xi >= 0 && xi < WDIM && yi >= 0 && yi < HDIM && zi >= 0 && zi < DDIM) {
                    size_t p = (((size_t)zi * HDIM + yi) * WDIM + xi) * 4;
                    o4[0] += wgt * (float)fb[p + 0];
                    o4[1] += wgt * (float)fb[p + 1];
                    o4[2] += wgt * (float)fb[p + 2];
                    o4[3] += wgt * (float)fb[p + 3];
                }
            }
}

__global__ void motion_kernel(const bf16* __restrict__ f, const float* __restrict__ kpb,
                              float* __restrict__ sm, bf16* __restrict__ inp)
{
    long idx = (long)blockIdx.x * 256 + threadIdx.x;
    if (idx >= N_FULL) return;
    int w = (int)(idx & 63), h = (int)((idx >> 6) & 63);
    int d = (int)((idx >> 12) & 15), b = (int)(idx >> 16);
    float gx = 2.f * w / 63.f - 1.f;
    float gy = 2.f * h / 63.f - 1.f;
    float gz = 2.f * d / 15.f - 1.f;

    const float* ks = kpb + b * 3;
    const float* kd = kpb + 24 + b * 3;
    const float* J  = kpb + 48 + b * 9;

    float dx = gx - kd[0], dy = gy - kd[1], dz = gz - kd[2];
    float sx = J[0]*dx + J[1]*dy + J[2]*dz + ks[0];
    float sy = J[3]*dx + J[4]*dy + J[5]*dz + ks[1];
    float sz = J[6]*dx + J[7]*dy + J[8]*dz + ks[2];

    long vox = idx & 0xFFFF;
    size_t s0 = ((size_t)b * 2 + 0) * DHW * 3 + (size_t)vox * 3;
    size_t s1 = s0 + (size_t)DHW * 3;
    sm[s0+0] = gx; sm[s0+1] = gy; sm[s0+2] = gz;
    sm[s1+0] = sx; sm[s1+1] = sy; sm[s1+2] = sz;

    float ex = gx - ks[0], ey = gy - ks[1], ez = gz - ks[2];
    float hm = __expf(-50.f * (dx*dx + dy*dy + dz*dz))
             - __expf(-50.f * (ex*ex + ey*ey + ez*ez));   // 0.5/0.01 = 50

    const bf16* fb = f + (size_t)b * DHW * 4;
    float v0[4], v1[4];
    trisample(fb, gx, gy, gz, v0);
    trisample(fb, sx, sy, sz, v1);

    bf16* o = inp + (size_t)idx * 10;
    o[0] = (bf16)0.f;
    o[1] = (bf16)v0[0]; o[2] = (bf16)v0[1]; o[3] = (bf16)v0[2]; o[4] = (bf16)v0[3];
    o[5] = (bf16)hm;
    o[6] = (bf16)v1[0]; o[7] = (bf16)v1[1]; o[8] = (bf16)v1[2]; o[9] = (bf16)v1[3];
}

// ======================================================================
// 2-way softmax mask + deformation reduction (writes final outputs).
// ======================================================================
__global__ void mask_deform_kernel(const bf16* __restrict__ mY, const float* __restrict__ mB,
                                   const float* __restrict__ sm,
                                   float* __restrict__ outD, float* __restrict__ outM)
{
    long idx = (long)blockIdx.x * 256 + threadIdx.x;
    if (idx >= N_FULL) return;
    int  b   = (int)(idx >> 16);
    long vox = idx & 0xFFFF;
    float m0 = (float)mY[(size_t)idx * 2 + 0] + mB[0];
    float m1 = (float)mY[(size_t)idx * 2 + 1] + mB[1];
    float mx = fmaxf(m0, m1);
    float e0 = __expf(m0 - mx), e1 = __expf(m1 - mx);
    float inv = 1.f / (e0 + e1);
    float p0 = e0 * inv, p1 = e1 * inv;
    outM[((size_t)b * 2 + 0) * DHW + vox] = p0;
    outM[((size_t)b * 2 + 1) * DHW + vox] = p1;
    const float* s0 = sm + ((size_t)b * 2) * DHW * 3 + (size_t)vox * 3;
    const float* s1 = s0 + (size_t)DHW * 3;
    float* od = outD + (size_t)idx * 3;
    od[0] = p0 * s0[0] + p1 * s1[0];
    od[1] = p0 * s0[1] + p1 * s1[1];
    od[2] = p0 * s0[2] + p1 * s1[2];
}

// ======================================================================
// Occlusion: 2-D 7x7 conv over (42*16)-channel reshape of prediction,
// + sigmoid.  ~1 GMAC total — plain VALU is fine.
// ======================================================================
__global__ void occl_kernel(const bf16* __restrict__ pred, const float* __restrict__ ow,
                            const float* __restrict__ ob, float* __restrict__ outO)
{
    int idx = blockIdx.x * 256 + threadIdx.x;
    if (idx >= BSZ * HDIM * WDIM) return;
    int w = idx & 63, h = (idx >> 6) & 63, b = idx >> 12;
    float acc = ob[0];
    for (int c = 0; c < 42; ++c)
        for (int d = 0; d < 16; ++d) {
            const float* wr = ow + (size_t)(c * 16 + d) * 49;
            for (int kh = 0; kh < 7; ++kh) {
                int ih = h + kh - 3;
                if (ih < 0 || ih >= HDIM) continue;
                for (int kw = 0; kw < 7; ++kw) {
                    int iw = w + kw - 3;
                    if (iw < 0 || iw >= WDIM) continue;
                    size_t n = (((size_t)b * DDIM + d) * HDIM + ih) * WDIM + iw;
                    acc += wr[kh * 7 + kw] * (float)pred[n * 42 + c];
                }
            }
        }
    outO[idx] = 1.f / (1.f + __expf(-acc));
}

// ======================================================================
// Host driver
// ======================================================================
static inline int  r32i(int x)         { return (x + 31) & ~31; }
static inline int  r64i(int x)         { return (x + 63) & ~63; }
static inline long cdiv(long a, long b){ return (a + b - 1) / b; }

extern "C" void kernel_launch(void* const* d_in, const int* in_sizes, int n_in,
                              void* d_out, int out_size, void* d_ws, size_t ws_size,
                              hipStream_t stream)
{
    (void)in_sizes; (void)n_in; (void)out_size; (void)ws_size;
    auto p = [&](int i) { return (const float*)d_in[i]; };

    // ---- workspace bump allocator ----
    char* base = (char*)d_ws;
    size_t off = 0;
    auto alloc = [&](size_t bytes) -> char* {
        char* r = base + off;
        off = (off + bytes + 255) & ~(size_t)255;
        return r;
    };

    long NL[6];                       // voxels at each H,W level
    for (int l = 0; l < 6; ++l) NL[l] = (long)BSZ * DDIM * (HDIM >> l) * (WDIM >> l);

    float* kpbuf  = (float*)alloc(256 * sizeof(float));
    float* stats  = (float*)alloc(2048 * sizeof(float));
    bf16*  featCL = (bf16*)alloc((size_t)N_FULL * 32 * 2);
    bf16*  convY  = (bf16*)alloc((size_t)N_FULL * 64 * 2);  // reused conv out, max 64ch full-res
    bf16*  fbuf   = (bf16*)alloc((size_t)N_FULL * 4 * 2);
    bf16*  inp    = (bf16*)alloc((size_t)N_FULL * 10 * 2);
    float* sm     = (float*)alloc((size_t)BSZ * 2 * DHW * 3 * 4);

    const int dnCi[5] = {10, 64, 128, 256, 512};
    const int dnCo[5] = {64, 128, 256, 512, 1024};
    const int upCi[5] = {128, 256, 512, 1024, 1024};  // index by i (up_i)
    const int upCo[5] = {32, 64, 128, 256, 512};

    bf16* pool[5];
    for (int i = 0; i < 5; ++i) pool[i] = (bf16*)alloc((size_t)NL[i + 1] * dnCo[i] * 2);
    bf16* cat1 = (bf16*)alloc((size_t)NL[1] * 128 * 2);
    bf16* cat2 = (bf16*)alloc((size_t)NL[2] * 256 * 2);
    bf16* cat3 = (bf16*)alloc((size_t)NL[3] * 512 * 2);
    bf16* cat4 = (bf16*)alloc((size_t)NL[4] * 1024 * 2);
    bf16* pred = (bf16*)alloc((size_t)N_FULL * 42 * 2);

    auto wBytes = [&](int Co, int Ci, int KS) {
        return (size_t)r32i(Co) * r64i(Ci * KS * KS * KS) * 2;
    };
    bf16* WpC = (bf16*)alloc(wBytes(4, 32, 1));
    bf16* WpD[5]; for (int i = 0; i < 5; ++i) WpD[i] = (bf16*)alloc(wBytes(dnCo[i], dnCi[i], 3));
    bf16* WpU[5]; for (int i = 0; i < 5; ++i) WpU[i] = (bf16*)alloc(wBytes(upCo[i], upCi[i], 3));
    bf16* WpM = (bf16*)alloc(wBytes(2, 42, 7));

    // ---- helpers ----
    auto packW = [&](const float* w, bf16* wp, int Co, int Ci, int KS) {
        int CoPad = r32i(Co), Ktot = Ci * KS * KS * KS, Kpad = r64i(Ktot);
        long total = (long)CoPad * Kpad;
        repack_w_kernel<<<(unsigned)cdiv(total, 256), 256, 0, stream>>>(w, wp, Co, Ci, KS, CoPad, Kpad, Ktot);
    };
    auto runConv = [&](const bf16* X, const bf16* Wp_, bf16* Y, bool useStats,
                       int D, int H, int W, int Ci, int Co, int KS, int ups) {
        int CoPad = r32i(Co), Ktot = Ci * KS * KS * KS, Kpad = r64i(Ktot);
        if (useStats)
            zero_f32_kernel<<<(unsigned)cdiv(2 * CoPad, 256), 256, 0, stream>>>(stats, 2 * CoPad);
        long Ntot = (long)BSZ * D * H * W;
        dim3 g((unsigned)(Ntot / 128), (unsigned)(CoPad / 32));
        float* st = useStats ? stats : nullptr;
        int spStep = 64 / Ci, ciStep = 64 % Ci;
        switch (KS) {
        case 1:
            conv3d_wmma_kernel<1><<<g, 256, 0, stream>>>(X, Wp_, Y, st, D, H, W, Ci, Co, CoPad,
                                                         Kpad, ups, spStep, ciStep);
            break;
        case 3:
            conv3d_wmma_kernel<3><<<g, 256, 0, stream>>>(X, Wp_, Y, st, D, H, W, Ci, Co, CoPad,
                                                         Kpad, ups, spStep, ciStep);
            break;
        default:
            conv3d_wmma_kernel<7><<<g, 256, 0, stream>>>(X, Wp_, Y, st, D, H, W, Ci, Co, CoPad,
                                                         Kpad, ups, spStep, ciStep);
            break;
        }
    };
    auto runBN = [&](const bf16* Yc, const float* g, const float* be, bf16* dst,
                     int D, int H, int W, int Co, int Cdst, int coff, int pl) {
        long total = (long)BSZ * D * (pl ? H / 2 : H) * (pl ? W / 2 : W) * Co;
        bn_apply_kernel<<<(unsigned)cdiv(total, 256), 256, 0, stream>>>(
            Yc, stats, g, be, dst, BSZ, D, H, W, Co, r32i(Co), Cdst, coff, pl);
    };
    auto runCopy = [&](const bf16* src, bf16* dst, long N, int Csrc, int Cdst, int coff) {
        long total = N * Csrc;
        copy_ch_kernel<<<(unsigned)cdiv(total, 256), 256, 0, stream>>>(src, dst, N, Csrc, Cdst, coff);
    };

    // ---- pack all conv weights to bf16 WMMA layout ----
    packW(p(9),  WpC, 4, 32, 1);
    for (int i = 0; i < 5; ++i) packW(p(13 + 4 * i), WpD[i], dnCo[i], dnCi[i], 3);
    // up params inserted i=4..0: up4 @33, up3 @37, up2 @41, up1 @45, up0 @49
    for (int i = 4; i >= 0; --i) packW(p(33 + 4 * (4 - i)), WpU[i], upCo[i], upCi[i], 3);
    packW(p(53), WpM, 2, 42, 7);

    // ---- feature -> channel-last bf16 ----
    {
        long total = N_FULL * 32;
        to_cl_bf16_kernel<<<(unsigned)cdiv(total, 256), 256, 0, stream>>>(p(0), featCL, BSZ, 32, (long)DHW);
    }

    // ---- compress conv (1x1x1, 32->4) + BN + ReLU ----
    runConv(featCL, WpC, convY, true, DDIM, HDIM, WDIM, 32, 4, 1, 0);
    runBN(convY, p(11), p(12), fbuf, DDIM, HDIM, WDIM, 4, 4, 0, 0);

    // ---- keypoints / jacobian + sparse motion + heatmap + grid sample ----
    kp_setup_kernel<<<1, 32, 0, stream>>>(p(1), p(2), p(3), p(4), p(6), p(7), p(8), kpbuf);
    motion_kernel<<<(unsigned)cdiv(N_FULL, 256), 256, 0, stream>>>(fbuf, kpbuf, sm, inp);

    // ---- down path: conv3x3 -> BN+ReLU -> avgpool(1,2,2) ----
    for (int i = 0; i < 5; ++i) {
        const bf16* X = (i == 0) ? inp : pool[i - 1];
        int H = HDIM >> i, W = WDIM >> i;
        runConv(X, WpD[i], convY, true, DDIM, H, W, dnCi[i], dnCo[i], 3, 0);
        runBN(convY, p(15 + 4 * i), p(16 + 4 * i), pool[i], DDIM, H, W, dnCo[i], dnCo[i], 0, 1);
    }

    // ---- skip copies into concat buffers ----
    runCopy(pool[3], cat4, NL[4], 512, 1024, 512);
    runCopy(pool[2], cat3, NL[3], 256, 512, 256);
    runCopy(pool[1], cat2, NL[2], 128, 256, 128);
    runCopy(pool[0], cat1, NL[1], 64, 128, 64);
    runCopy(inp,     pred, NL[0], 10, 42, 32);

    // ---- up path: upsample(folded) -> conv3x3 -> BN+ReLU -> concat ----
    const bf16* upX[5]  = {cat1, cat2, cat3, cat4, pool[4]};  // input of up_i
    bf16*       upD[5]  = {pred, cat1, cat2, cat3, cat4};     // dest of up_i
    const int   upCd[5] = {42, 128, 256, 512, 1024};
    for (int i = 4; i >= 0; --i) {
        int H = HDIM >> i, W = WDIM >> i;              // conv output resolution
        int gi = 35 + 4 * (4 - i), bi = gi + 1;        // up_i gamma/beta param index
        runConv(upX[i], WpU[i], convY, true, DDIM, H, W, upCi[i], upCo[i], 3, 1);
        runBN(convY, p(gi), p(bi), upD[i], DDIM, H, W, upCo[i], upCd[i], 0, 0);
    }

    // ---- mask conv (7x7x7, 42->2, no BN) ----
    runConv(pred, WpM, convY, false, DDIM, HDIM, WDIM, 42, 2, 7, 0);

    // ---- final outputs ----
    float* outD = (float*)d_out;
    float* outM = outD + (size_t)N_FULL * 3;
    float* outO = outM + (size_t)BSZ * 2 * DHW;
    mask_deform_kernel<<<(unsigned)cdiv(N_FULL, 256), 256, 0, stream>>>(convY, p(54), sm, outD, outM);
    occl_kernel<<<(unsigned)cdiv((long)BSZ * HDIM * WDIM, 256), 256, 0, stream>>>(pred, p(55), p(56), outO);
}
